// MultiHeadLocalAttention_65240553226622
// MI455X (gfx1250) — compile-verified, matrix operand order checked by reading, not run on hardware
//
#include <hip/hip_runtime.h>
#include <hip/hip_bf16.h>

typedef __attribute__((ext_vector_type(16))) _Float16 v16h;
typedef __attribute__((ext_vector_type(8)))  _Float16 v8h;
typedef __attribute__((ext_vector_type(8)))  float    v8f;
typedef __attribute__((ext_vector_type(4)))  unsigned int v4u;
typedef __attribute__((ext_vector_type(8)))  int      v8i;
typedef __attribute__((ext_vector_type(4)))  int      v4i;

#define E_    384
#define H_    6
#define D_    64
#define L_    1025
#define B_    8
#define MTOT  (B_*L_)     /* 8200 rows */
#define BH_   (B_*H_)     /* 48 */
#define WELEM (E_*E_)     /* 147456 */

// ---------------------------------------------------------------------------
// WMMA helpers (CDNA5 v_wmma_f32_16x16x32_f16, wave32)
// ---------------------------------------------------------------------------
__device__ __forceinline__ v8f wmma_f16(v16h a, v16h b, v8f c) {
  return __builtin_amdgcn_wmma_f32_16x16x32_f16(false, a, false, b,
                                                (short)0, c, false, false);
}

__device__ __forceinline__ v16h frag_ld(const _Float16* base, int stride, int lane) {
  int m = lane & 15, hf = lane >> 4;
  const _Float16* r = base + m * stride + 8 * hf;
  v8h lo = *(const v8h*)(r);
  v8h hi = *(const v8h*)(r + 16);
  v16h f;
#pragma unroll
  for (int i = 0; i < 8; ++i) { f[i] = lo[i]; f[i + 8] = hi[i]; }
  return f;
}

__device__ __forceinline__ v16h frag_ld_row(const _Float16* rowbase, int hf) {
  const _Float16* r = rowbase + 8 * hf;
  v8h lo = *(const v8h*)(r);
  v8h hi = *(const v8h*)(r + 16);
  v16h f;
#pragma unroll
  for (int i = 0; i < 8; ++i) { f[i] = lo[i]; f[i + 8] = hi[i]; }
  return f;
}

// ---------------------------------------------------------------------------
// Tensor Data Mover: async strided tile load (global -> LDS), 2-byte elems.
// D# layout per CDNA5 ISA ch.8: group0 (count=1, lds_addr, global_addr,
// type=2), group1 (data_size=2B, dims/strides), groups 2-3 (dim2 / unused).
// Walk order: x = tile_d0 (contiguous halfs), y = rows (stride s0 elems),
// z = matrices (stride s1 elems).  Issued per-wave; tracked by TENSORcnt.
// ---------------------------------------------------------------------------
__device__ __forceinline__ void tdm_load_tile_f16(
    unsigned lds_addr, unsigned long long gaddr_bytes,
    unsigned tile_d0, unsigned tile_d1, unsigned tile_d2,
    unsigned long long s0_elems, unsigned long long s1_elems) {
  const unsigned td0 = 0x100000u, td1 = 0x100000u;   // huge dims: no OOB clip
  v4u g0;
  g0[0] = 1u;                                        // count=1, user D#
  g0[1] = lds_addr;                                  // bits 63:32
  g0[2] = (unsigned)(gaddr_bytes & 0xffffffffull);   // global_addr[31:0]
  g0[3] = (unsigned)((gaddr_bytes >> 32) & 0x1ffffffull) | 0x80000000u; // type=2
  v8i g1;
  g1[0] = (int)(1u << 16);                           // data_size = 1 (2 bytes)
  g1[1] = (int)((td0 & 0xffffu) << 16);              // tensor_dim0[15:0]
  g1[2] = (int)(((td0 >> 16) & 0xffffu) | ((td1 & 0xffffu) << 16));
  g1[3] = (int)(((td1 >> 16) & 0xffffu) | ((tile_d0 & 0xffffu) << 16));
  g1[4] = (int)((tile_d1 & 0xffffu) | ((tile_d2 & 0xffffu) << 16));
  g1[5] = (int)(unsigned)(s0_elems & 0xffffffffull); // dim0_stride[31:0]
  g1[6] = (int)((unsigned)((s0_elems >> 32) & 0xffffull) |
                (unsigned)((s1_elems & 0xffffull) << 16));
  g1[7] = (int)(unsigned)((s1_elems >> 16) & 0xffffffffull);
  v4i g2;
  g2[0] = 16; g2[1] = 0; g2[2] = 0; g2[3] = 0;       // tensor_dim2 (>= tile_d2)
  v4i g3;
  g3[0] = 0; g3[1] = 0; g3[2] = 0; g3[3] = 0;
#if defined(__clang_major__) && (__clang_major__ >= 23)
  v8i g4; g4[0]=0; g4[1]=0; g4[2]=0; g4[3]=0; g4[4]=0; g4[5]=0; g4[6]=0; g4[7]=0;
  __builtin_amdgcn_tensor_load_to_lds(g0, g1, g2, g3, g4, 0);
#else
  __builtin_amdgcn_tensor_load_to_lds(g0, g1, g2, g3, 0);
#endif
}

// ---------------------------------------------------------------------------
// Kernel 0: convert 4 weight matrices f32 -> f16, transposed to [n][k]
// ---------------------------------------------------------------------------
__global__ void convert_w(const float* __restrict__ Wq, const float* __restrict__ Wk,
                          const float* __restrict__ Wv, const float* __restrict__ Wo,
                          _Float16* __restrict__ WT) {
  int idx = blockIdx.x * blockDim.x + threadIdx.x;
  if (idx >= 4 * WELEM) return;
  int mat = idx / WELEM;
  int r   = idx - mat * WELEM;
  int n   = r / E_;
  int k   = r - n * E_;
  const float* W = (mat == 0) ? Wq : (mat == 1) ? Wk : (mat == 2) ? Wv : Wo;
  WT[idx] = (_Float16)W[k * E_ + n];   // WT[mat][n][k] = W[k][n]
}

// ---------------------------------------------------------------------------
// Kernel 1: fused QKV projection.  grid = (129, 6); block = 256 (8 waves).
// Weight tiles ([3][64][32] f16) arrive via one 3D TDM load per k-step.
// ---------------------------------------------------------------------------
__global__ __launch_bounds__(256) void qkv_proj(
    const float* __restrict__ x,
    const _Float16* __restrict__ WqT,   // contiguous: WqT, +WELEM = Wk, +2*WELEM = Wv
    const float* __restrict__ bq, const float* __restrict__ bk,
    const float* __restrict__ bv,
    _Float16* __restrict__ Qh, _Float16* __restrict__ Kh, _Float16* __restrict__ Vh) {
  __shared__ __align__(16) _Float16 xa[64 * 32];
  __shared__ __align__(16) _Float16 wb[3][64 * 32];

  const int tid = threadIdx.x, lane = tid & 31, wv = tid >> 5;
  const int wr = wv >> 1, wc = wv & 1;
  const int mt = blockIdx.x, h = blockIdx.y;

  const float* Bs[3] = {bq, bk, bv};
  _Float16*    Out[3];
  Out[0] = Qh; Out[1] = Kh; Out[2] = Vh;

  v8f acc[3][2];
#pragma unroll
  for (int j = 0; j < 3; ++j)
#pragma unroll
    for (int s = 0; s < 2; ++s) acc[j][s] = v8f{0,0,0,0,0,0,0,0};

  const int frow = tid >> 2;          // 0..63: LDS fill row
  const int fk   = (tid & 3) * 8;     // 0,8,16,24: k chunk
  const int m_g0 = mt * 64 + frow;
  const int m_cl = (m_g0 < MTOT) ? m_g0 : (MTOT - 1);

  const unsigned wb_lds = (unsigned)(unsigned long long)(uintptr_t)&wb[0][0];
  const unsigned long long wbase =
      (unsigned long long)(uintptr_t)WqT + (unsigned long long)(h * 64) * E_ * 2ull;

  for (int kk = 0; kk < E_; kk += 32) {
    // async DMA of the three weight tiles: tile 32(k) x 64(n) x 3(mat)
    if (wv == 0) {
      tdm_load_tile_f16(wb_lds, wbase + (unsigned long long)kk * 2ull,
                        /*tile_d0=*/32, /*tile_d1=*/64, /*tile_d2=*/3,
                        /*s0=*/E_, /*s1=*/WELEM);
    }
    // stage x tile (f32 -> f16) while the DMA runs
    const float4* xs4 = (const float4*)(x + (size_t)m_cl * E_ + kk + fk);
    float4 u0 = xs4[0], u1 = xs4[1];
    v8h hv;
    hv[0]=(_Float16)u0.x; hv[1]=(_Float16)u0.y; hv[2]=(_Float16)u0.z; hv[3]=(_Float16)u0.w;
    hv[4]=(_Float16)u1.x; hv[5]=(_Float16)u1.y; hv[6]=(_Float16)u1.z; hv[7]=(_Float16)u1.w;
    *(v8h*)&xa[frow * 32 + fk] = hv;
    if (kk + 32 < E_) __builtin_prefetch(x + (size_t)m_cl * E_ + kk + 32 + fk, 0, 3);
    if (wv == 0) __builtin_amdgcn_s_wait_tensorcnt(0);
    __syncthreads();

    v16h a = frag_ld(&xa[(wr * 16) * 32], 32, lane);
#pragma unroll
    for (int j = 0; j < 3; ++j)
#pragma unroll
      for (int s = 0; s < 2; ++s) {
        v16h b = frag_ld(&wb[j][(wc * 32 + s * 16) * 32], 32, lane);
        acc[j][s] = wmma_f16(a, b, acc[j][s]);
      }
    __syncthreads();
  }

  // store (B,H,L,D) f16 with bias.  C layout: lane n=lane&15, rows 8*half+r.
  const int hf = lane >> 4, n = lane & 15;
#pragma unroll
  for (int j = 0; j < 3; ++j)
#pragma unroll
    for (int s = 0; s < 2; ++s) {
      const int d    = wc * 32 + s * 16 + n;   // 0..63 within head
      const float bi = Bs[j][h * 64 + d];
#pragma unroll
      for (int r = 0; r < 8; ++r) {
        int m_g = mt * 64 + wr * 16 + 8 * hf + r;
        if (m_g < MTOT) {
          int b = m_g / L_;
          int l = m_g - b * L_;
          Out[j][(((size_t)b * H_ + h) * L_ + l) * D_ + d] = (_Float16)(acc[j][s][r] + bi);
        }
      }
    }
}

// ---------------------------------------------------------------------------
// Kernel 2a: CLS query (row 0 of each (b,h)) full attention.  grid = 48.
// ---------------------------------------------------------------------------
__global__ __launch_bounds__(256) void cls_attn(
    const _Float16* __restrict__ Qh, const _Float16* __restrict__ Kh,
    const _Float16* __restrict__ Vh, _Float16* __restrict__ Ah) {
  __shared__ float sq[D_];
  __shared__ float sw[L_ + 31];
  __shared__ float red[256];
  __shared__ float part[4][D_];

  const int bh = blockIdx.x, tid = threadIdx.x;
  const _Float16* Qp = Qh + (size_t)bh * L_ * D_;
  const _Float16* Kp = Kh + (size_t)bh * L_ * D_;
  const _Float16* Vp = Vh + (size_t)bh * L_ * D_;

  if (tid < D_) sq[tid] = (float)Qp[tid];
  __syncthreads();

  float lmax = -3.0e38f;
  for (int k = tid; k < L_; k += 256) {
    const _Float16* kr = Kp + (size_t)k * D_;
    float s = 0.f;
#pragma unroll
    for (int d = 0; d < D_; ++d) s += sq[d] * (float)kr[d];
    s *= 0.125f;
    sw[k] = s;
    lmax = fmaxf(lmax, s);
  }
  red[tid] = lmax; __syncthreads();
  for (int st = 128; st > 0; st >>= 1) {
    if (tid < st) red[tid] = fmaxf(red[tid], red[tid + st]);
    __syncthreads();
  }
  const float mx = red[0];
  __syncthreads();

  float lsum = 0.f;
  for (int k = tid; k < L_; k += 256) {
    float p = exp2f((sw[k] - mx) * 1.44269504f);
    sw[k] = p;
    lsum += p;
  }
  red[tid] = lsum; __syncthreads();
  for (int st = 128; st > 0; st >>= 1) {
    if (tid < st) red[tid] += red[tid + st];
    __syncthreads();
  }
  const float inv = 1.0f / red[0];

  const int d = tid & 63, c = tid >> 6;
  float acc = 0.f;
  for (int k = c; k < L_; k += 4) acc += sw[k] * (float)Vp[(size_t)k * D_ + d];
  part[c][d] = acc; __syncthreads();
  if (tid < D_) {
    float o = (part[0][tid] + part[1][tid] + part[2][tid] + part[3][tid]) * inv;
    Ah[(size_t)bh * L_ * D_ + tid] = (_Float16)o;
  }
}

// ---------------------------------------------------------------------------
// Kernel 2b: windowed attention.  3072 blocks of 16 queries; 4 waves/WG,
// one query-block per wave.  grid = 768, block = 128.
// Slot w in [0,64): w<48 -> key qs-16+w; w==48 -> CLS; w>48 -> padding.
// ---------------------------------------------------------------------------
__global__ __launch_bounds__(128) void win_attn(
    const _Float16* __restrict__ Qh, const _Float16* __restrict__ Kh,
    const _Float16* __restrict__ Vh, _Float16* __restrict__ Ah) {
  __shared__ __align__(16) _Float16 lds_p[4][16 * 64];
  __shared__ __align__(16) _Float16 lds_v[4][64 * 64];

  const int tid  = threadIdx.x;
  const int lane = tid & 31;
  const int wv   = tid >> 5;
  const int gblk = blockIdx.x * 4 + wv;   // 0..3071
  const int bh   = gblk >> 6;             // (b*H + h)
  const int qb   = gblk & 63;
  const int qs   = 1 + qb * 16;           // first query of this block

  const _Float16* Qb = Qh + ((size_t)bh * L_ + qs) * D_;
  const _Float16* Kb = Kh + (size_t)bh * L_ * D_;
  const _Float16* Vb = Vh + (size_t)bh * L_ * D_;

  const int hf = lane >> 4, n = lane & 15;

  // Q fragments, straight from global (row-major, stride 64)
  v16h aq0 = frag_ld(Qb + 0, D_, lane);
  v16h aq1 = frag_ld(Qb + 32, D_, lane);

  // Stage V window transposed into LDS: lds_v[d][w]
  for (int w = lane; w < 64; w += 32) {
    int idx = (w == 48) ? 0 : min(max(qs - 16 + w, 0), L_ - 1);
    const _Float16* src = Vb + (size_t)idx * D_;
#pragma unroll
    for (int d = 0; d < D_; ++d) lds_v[wv][d * 64 + w] = src[d];
  }

  // Scores: S(16q x 64w) = Q(16x64) * Kw^T; K rows are natural [n=slot][k=d]
  v8f sacc[4];
#pragma unroll
  for (int t = 0; t < 4; ++t) {
    sacc[t] = v8f{0,0,0,0,0,0,0,0};
    int w   = 16 * t + n;
    int idx = (w == 48) ? 0 : min(max(qs - 16 + w, 0), L_ - 1);
    const _Float16* kr = Kb + (size_t)idx * D_;
    v16h b0 = frag_ld_row(kr + 0, hf);
    v16h b1 = frag_ld_row(kr + 32, hf);
    sacc[t] = wmma_f16(aq0, b0, sacc[t]);
    sacc[t] = wmma_f16(aq1, b1, sacc[t]);
  }

  // Band mask + row softmax (rows live in 16-lane halves; xor-reduce <16)
  float rowsum[8];
#pragma unroll
  for (int r = 0; r < 8; ++r) {
    const int m = 8 * hf + r;          // query row in block
    float mx = -3.0e38f;
#pragma unroll
    for (int t = 0; t < 4; ++t) {
      int w = 16 * t + n;
      int idx = qs - 16 + w;
      bool valid = (w == 48) ||
                   (w < 48 && w >= m && w <= m + 32 && idx >= 1 && idx <= L_ - 1);
      float s = valid ? sacc[t][r] * 0.125f : -3.0e38f;
      sacc[t][r] = s;
      mx = fmaxf(mx, s);
    }
#pragma unroll
    for (int o = 1; o < 16; o <<= 1) mx = fmaxf(mx, __shfl_xor(mx, o));
    float sum = 0.f;
#pragma unroll
    for (int t = 0; t < 4; ++t) {
      float p = exp2f((sacc[t][r] - mx) * 1.44269504f);  // underflows to 0 if masked
      sacc[t][r] = p;
      sum += p;
    }
#pragma unroll
    for (int o = 1; o < 16; o <<= 1) sum += __shfl_xor(sum, o);
    rowsum[r] = sum;
  }

  // P (C layout) -> LDS [q][w] so it can be re-read as an A fragment
#pragma unroll
  for (int r = 0; r < 8; ++r)
#pragma unroll
    for (int t = 0; t < 4; ++t)
      lds_p[wv][(8 * hf + r) * 64 + 16 * t + n] = (_Float16)sacc[t][r];
  __syncthreads();

  // O(16q x 64d) = P(16x64w) * Vw(64w x 64d)
  v16h pa0 = frag_ld(&lds_p[wv][0], 64, lane);
  v16h pa1 = frag_ld(&lds_p[wv][32], 64, lane);
  v8f oacc[4];
#pragma unroll
  for (int t2 = 0; t2 < 4; ++t2) {
    oacc[t2] = v8f{0,0,0,0,0,0,0,0};
    v16h vb0 = frag_ld(&lds_v[wv][(16 * t2) * 64 + 0], 64, lane);
    v16h vb1 = frag_ld(&lds_v[wv][(16 * t2) * 64 + 32], 64, lane);
    oacc[t2] = wmma_f16(pa0, vb0, oacc[t2]);
    oacc[t2] = wmma_f16(pa1, vb1, oacc[t2]);
  }

  // normalize + store to (B,H,L,D) f16
#pragma unroll
  for (int t2 = 0; t2 < 4; ++t2) {
    const int d = 16 * t2 + n;
#pragma unroll
    for (int r = 0; r < 8; ++r) {
      const int m = 8 * hf + r;
      float o = oacc[t2][r] / rowsum[r];
      Ah[((size_t)bh * L_ + qs + m) * D_ + d] = (_Float16)o;
    }
  }
}

// ---------------------------------------------------------------------------
// Kernel 3: output projection.  grid = (129, 6); block = 256 (8 waves).
// Wo tile (64x32 f16, row stride 384) arrives via one 2D TDM load per k-step.
// ---------------------------------------------------------------------------
__global__ __launch_bounds__(256) void out_proj(
    const _Float16* __restrict__ Ah, const _Float16* __restrict__ WoT,
    const float* __restrict__ bo, float* __restrict__ outp) {
  __shared__ __align__(16) _Float16 xa[64 * 32];
  __shared__ __align__(16) _Float16 wb[64 * 32];

  const int tid = threadIdx.x, lane = tid & 31, wv = tid >> 5;
  const int wr = wv >> 1, wc = wv & 1;
  const int mt = blockIdx.x, nt = blockIdx.y;

  v8f acc[2];
  acc[0] = v8f{0,0,0,0,0,0,0,0};
  acc[1] = v8f{0,0,0,0,0,0,0,0};

  const int frow = tid >> 2;
  const int fk   = (tid & 3) * 8;
  const int m_g0 = mt * 64 + frow;
  const int m_cl = (m_g0 < MTOT) ? m_g0 : (MTOT - 1);
  const int bb   = m_cl / L_;
  const int ll   = m_cl - bb * L_;

  const unsigned wb_lds = (unsigned)(unsigned long long)(uintptr_t)&wb[0];
  const unsigned long long wbase =
      (unsigned long long)(uintptr_t)WoT + (unsigned long long)(nt * 64) * E_ * 2ull;

  for (int kk = 0; kk < E_; kk += 32) {
    if (wv == 0) {
      tdm_load_tile_f16(wb_lds, wbase + (unsigned long long)kk * 2ull,
                        /*tile_d0=*/32, /*tile_d1=*/64, /*tile_d2=*/0,
                        /*s0=*/E_, /*s1=*/WELEM);
    }
    const int k0 = kk + fk;
    const int h2 = k0 >> 6, d0 = k0 & 63;
    const _Float16* asrc = Ah + (((size_t)bb * H_ + h2) * L_ + ll) * D_ + d0;
    *(v8h*)&xa[frow * 32 + fk] = *(const v8h*)asrc;
    if (wv == 0) __builtin_amdgcn_s_wait_tensorcnt(0);
    __syncthreads();

    v16h a = frag_ld(&xa[(wr * 16) * 32], 32, lane);
#pragma unroll
    for (int s = 0; s < 2; ++s) {
      v16h b = frag_ld(&wb[(wc * 32 + s * 16) * 32], 32, lane);
      acc[s] = wmma_f16(a, b, acc[s]);
    }
    __syncthreads();
  }

  const int hf = lane >> 4, n = lane & 15;
#pragma unroll
  for (int s = 0; s < 2; ++s) {
    const int n_g = nt * 64 + wc * 32 + s * 16 + n;
    const float bi = bo[n_g];
#pragma unroll
    for (int r = 0; r < 8; ++r) {
      int m_g = mt * 64 + wr * 16 + 8 * hf + r;
      if (m_g < MTOT) outp[(size_t)m_g * E_ + n_g] = acc[s][r] + bi;
    }
  }
}

// ---------------------------------------------------------------------------
// Host-side launcher
// ---------------------------------------------------------------------------
extern "C" void kernel_launch(void* const* d_in, const int* in_sizes, int n_in,
                              void* d_out, int out_size, void* d_ws, size_t ws_size,
                              hipStream_t stream) {
  (void)in_sizes; (void)n_in; (void)out_size; (void)ws_size;
  const float* x  = (const float*)d_in[0];
  const float* Wq = (const float*)d_in[1];
  const float* bq = (const float*)d_in[2];
  const float* Wk = (const float*)d_in[3];
  const float* bk = (const float*)d_in[4];
  const float* Wv = (const float*)d_in[5];
  const float* bv = (const float*)d_in[6];
  const float* Wo = (const float*)d_in[7];
  const float* bo = (const float*)d_in[8];
  float* outp = (float*)d_out;

  char* ws = (char*)d_ws;
  const size_t qkv_halfs = (size_t)BH_ * L_ * D_;          // 3,148,800
  const size_t qkv_bytes = qkv_halfs * sizeof(_Float16);   // 6,297,600
  size_t off = 0;
  _Float16* WT = (_Float16*)(ws + off); off += (size_t)4 * WELEM * sizeof(_Float16) + 4096;
  _Float16* Qh = (_Float16*)(ws + off); off += qkv_bytes + 4096;
  _Float16* Kh = (_Float16*)(ws + off); off += qkv_bytes + 4096;
  _Float16* Vh = (_Float16*)(ws + off); off += qkv_bytes + 4096;
  _Float16* Ah = (_Float16*)(ws + off); off += qkv_bytes + 4096;

  convert_w<<<(4 * WELEM + 255) / 256, 256, 0, stream>>>(Wq, Wk, Wv, Wo, WT);
  qkv_proj<<<dim3(129, 6), 256, 0, stream>>>(x, WT, bq, bk, bv, Qh, Kh, Vh);
  cls_attn<<<BH_, 256, 0, stream>>>(Qh, Kh, Vh, Ah);
  win_attn<<<768, 128, 0, stream>>>(Qh, Kh, Vh, Ah);
  out_proj<<<dim3(129, 6), 256, 0, stream>>>(Ah, WT + 3 * WELEM, bo, outp);
}
